// NewsConv_71502615544009
// MI455X (gfx1250) — compile-verified, matrix operand order checked by reading, not run on hardware
//
#include <hip/hip_runtime.h>
#include <hip/hip_bf16.h>
#include <cstdint>

// ---------------------------------------------------------------------------
// NewsConv on MI455X (gfx1250, wave32):
//   agg  = atomic f32 segment-mean of neighbor features (L2-resident gathers)
//   H    = tanh([news|ent|top] @ W1 + b1)   -> v_wmma_f32_16x16x32_bf16
//   out  = H @ W2 + b2                      -> v_wmma_f32_16x16x32_bf16
// Weights are pre-packed to bf16 WMMA B-fragment layout; A tiles are staged
// through LDS in fragment layout (ds_load_b128 in the inner loop).
// ---------------------------------------------------------------------------

typedef __attribute__((ext_vector_type(16))) __bf16 v16bf;
typedef __attribute__((ext_vector_type(8)))  __bf16 v8bf;
typedef __attribute__((ext_vector_type(8)))  float  v8f;

#define N_NEWS 100000
#define D      128
#define HID    512
#define OUTD   128
#define K1     384   // 3*D
#define KT1    12    // K1/32
#define KT2    16    // HID/32
#define NT1    32    // HID/16
#define NT2    8     // OUTD/16

static __device__ __forceinline__ __bf16 f2bf(float f) { return (__bf16)f; }

static __device__ __forceinline__ v16bf frag_from(v8bf lo, v8bf hi) {
  v16bf a;
#pragma unroll
  for (int j = 0; j < 8; ++j) { a[j] = lo[j]; a[8 + j] = hi[j]; }
  return a;
}

// ---------------------------------------------------------------- zero -----
__global__ void zero_f4(float4* __restrict__ p, long n4) {
  long i = (long)blockIdx.x * blockDim.x + threadIdx.x;
  long stride = (long)gridDim.x * blockDim.x;
  float4 z = {0.f, 0.f, 0.f, 0.f};
  for (; i < n4; i += stride) p[i] = z;
}

// ------------------------------------------------------- edge aggregation --
// One wave32 per edge: lane L moves floats [4L, 4L+4) of the 128-float row.
__global__ __launch_bounds__(256) void edge_agg(
    const float* __restrict__ feats, const int* __restrict__ row,
    const int* __restrict__ col, float* __restrict__ aggsum,  // [N_NEWS][256]
    float* __restrict__ deg, int nedges, int segoff) {
  const int lane = threadIdx.x & 31;
  long wid = ((long)blockIdx.x * blockDim.x + threadIdx.x) >> 5;
  long nw  = ((long)gridDim.x * blockDim.x) >> 5;
  for (long e = wid; e < nedges; e += nw) {
    int r = row[e], c = col[e];
    const float4 v = ((const float4*)(feats + (long)c * D))[lane];
    float* dst = aggsum + (long)r * 256 + segoff + lane * 4;
    atomicAdd(dst + 0, v.x);
    atomicAdd(dst + 1, v.y);
    atomicAdd(dst + 2, v.z);
    atomicAdd(dst + 3, v.w);
    if (lane == 0) atomicAdd(deg + r, 1.0f);
  }
}

// ------------------------------------------------------------ normalize ----
__global__ void normalize_agg(float* __restrict__ agg,
                              const float* __restrict__ dege,
                              const float* __restrict__ degt) {
  long i = (long)blockIdx.x * blockDim.x + threadIdx.x;
  long n = (long)N_NEWS * 256;
  long stride = (long)gridDim.x * blockDim.x;
  for (; i < n; i += stride) {
    long r = i >> 8;
    int  j = (int)(i & 255);
    float d = ((j < 128) ? dege[r] : degt[r]) + 1e-8f;
    agg[i] = agg[i] / d;
  }
}

// --------------------------------------------- weight B-fragment packing ---
// frag[ktile][ntile][lane][16]: lane(half=L>>4, n=L&15) holds
//   slots 0..7  = W[ktile*32 + 8*half + j     ][ntile*16 + n]
//   slots 8..15 = W[ktile*32 + 16 + 8*half + j][ntile*16 + n]
__global__ void make_wfrag(const float* __restrict__ W, __bf16* __restrict__ frag,
                           int Ncols, int nntiles, int nentries) {
  int idx = blockIdx.x * blockDim.x + threadIdx.x;
  if (idx >= nentries) return;
  int lane  = idx & 31;
  int ntile = (idx >> 5) % nntiles;
  int ktile = (idx >> 5) / nntiles;
  int half = lane >> 4, n = lane & 15;
  int k0 = ktile * 32 + 8 * half;
  int k1 = k0 + 16;
  int ncol = ntile * 16 + n;
  __bf16* dst = frag + (long)idx * 16;
#pragma unroll
  for (int j = 0; j < 8; ++j) {
    dst[j]     = f2bf(W[(long)(k0 + j) * Ncols + ncol]);
    dst[8 + j] = f2bf(W[(long)(k1 + j) * Ncols + ncol]);
  }
}

// ------------------------------------------------------------- GEMM 1 ------
// Block = 8 waves, one M-tile (16 rows). Phase 1: stage the 16x384 X tile as
// bf16 A-fragments in LDS. Phase 2: wave w computes cols [64w, 64w+64) as
// 4 WMMA N-tiles; A via ds_load_b128, B via global b128 from W1frag.
__global__ __launch_bounds__(256) void gemm1_tanh(
    const float* __restrict__ news,          // [N_NEWS][128]
    const float* __restrict__ agg,           // [N_NEWS][256] (ent|top)
    const __bf16* __restrict__ W1frag,       // [12][32][32][16]
    const float* __restrict__ b1,            // [512]
    __bf16* __restrict__ H16) {              // [N_NEWS][512] bf16
  __shared__ uint4 Alds[KT1 * 32 * 2];       // 12,288 B
  const int lane = threadIdx.x & 31;
  const int wv   = threadIdx.x >> 5;
  const int half = lane >> 4;
  const int n    = lane & 15;
  const int mtile = blockIdx.x;

  // ---- Phase 1: convert X tile -> A fragments in LDS (once per block) ----
  for (int e = threadIdx.x; e < KT1 * 32; e += 256) {
    const int kt = e >> 5, le = e & 31;
    const int hf = le >> 4, m = le & 15;
    const long row = (long)mtile * 16 + m;
    const int k0 = kt * 32 + 8 * hf;
    const int k1 = k0 + 16;
    const float* p0 = (k0 < 128) ? (news + row * D + k0)
                                 : (agg + row * 256 + (k0 - 128));
    const float* p1 = (k1 < 128) ? (news + row * D + k1)
                                 : (agg + row * 256 + (k1 - 128));
    float4 va0 = *(const float4*)(p0);
    float4 va1 = *(const float4*)(p0 + 4);
    float4 va2 = *(const float4*)(p1);
    float4 va3 = *(const float4*)(p1 + 4);
    v8bf lo, hi;
    lo[0]=f2bf(va0.x); lo[1]=f2bf(va0.y); lo[2]=f2bf(va0.z); lo[3]=f2bf(va0.w);
    lo[4]=f2bf(va1.x); lo[5]=f2bf(va1.y); lo[6]=f2bf(va1.z); lo[7]=f2bf(va1.w);
    hi[0]=f2bf(va2.x); hi[1]=f2bf(va2.y); hi[2]=f2bf(va2.z); hi[3]=f2bf(va2.w);
    hi[4]=f2bf(va3.x); hi[5]=f2bf(va3.y); hi[6]=f2bf(va3.z); hi[7]=f2bf(va3.w);
    Alds[e * 2 + 0] = __builtin_bit_cast(uint4, lo);
    Alds[e * 2 + 1] = __builtin_bit_cast(uint4, hi);
  }
  __syncthreads();

  // ---- Phase 2: WMMA ----
  v8f acc[4];
#pragma unroll
  for (int t = 0; t < 4; ++t) acc[t] = (v8f){0.f,0.f,0.f,0.f,0.f,0.f,0.f,0.f};

  for (int kt = 0; kt < KT1; ++kt) {
    const v16bf a = frag_from(
        __builtin_bit_cast(v8bf, Alds[(kt * 32 + lane) * 2 + 0]),
        __builtin_bit_cast(v8bf, Alds[(kt * 32 + lane) * 2 + 1]));
#pragma unroll
    for (int t = 0; t < 4; ++t) {
      const int ntile = wv * 4 + t;
      const uint4* bp = (const uint4*)(W1frag +
          ((long)(kt * NT1 + ntile) * 32 + lane) * 16);
      if (kt + 1 < KT1)
        __builtin_prefetch(W1frag + ((long)((kt + 1) * NT1 + ntile) * 32 + lane) * 16, 0, 1);
      const v16bf b = frag_from(__builtin_bit_cast(v8bf, bp[0]),
                                __builtin_bit_cast(v8bf, bp[1]));
      acc[t] = __builtin_amdgcn_wmma_f32_16x16x32_bf16(
          false, a, false, b, (short)0, acc[t], false, false);
    }
  }

  // Epilogue: bias + tanh, store bf16. C/D layout: VGPR r -> M = r + 8*half.
#pragma unroll
  for (int t = 0; t < 4; ++t) {
    const int ncol = wv * 64 + t * 16 + n;
    const float bias = b1[ncol];
#pragma unroll
    for (int r = 0; r < 8; ++r) {
      const long row = (long)mtile * 16 + r + 8 * half;
      H16[row * HID + ncol] = f2bf(tanhf(acc[t][r] + bias));
    }
  }
}

// ------------------------------------------------------------- GEMM 2 ------
// Block = 8 waves, one M-tile. Phase 1: stage the 16x512 bf16 H tile as
// fragments in LDS (pure b128 copies). Phase 2: wave w -> cols [16w, 16w+16).
__global__ __launch_bounds__(256) void gemm2_bias(
    const __bf16* __restrict__ H16,          // [N_NEWS][512] bf16
    const __bf16* __restrict__ W2frag,       // [16][8][32][16]
    const float* __restrict__ b2,            // [128]
    float* __restrict__ out) {               // [N_NEWS][128]
  __shared__ uint4 Hlds[KT2 * 32 * 2];       // 16,384 B
  const int lane = threadIdx.x & 31;
  const int wv   = threadIdx.x >> 5;
  const int half = lane >> 4;
  const int n    = lane & 15;
  const int mtile = blockIdx.x;
  const int ncol = wv * 16 + n;

  // ---- Phase 1: copy H tile into fragment-ordered LDS ----
  for (int e = threadIdx.x; e < KT2 * 32; e += 256) {
    const int kt = e >> 5, le = e & 31;
    const int hf = le >> 4, m = le & 15;
    const long row = (long)mtile * 16 + m;
    const int k0 = kt * 32 + 8 * hf;
    Hlds[e * 2 + 0] = *(const uint4*)(H16 + row * HID + k0);
    Hlds[e * 2 + 1] = *(const uint4*)(H16 + row * HID + k0 + 16);
  }
  __syncthreads();

  // ---- Phase 2: WMMA ----
  v8f acc = (v8f){0.f,0.f,0.f,0.f,0.f,0.f,0.f,0.f};
  for (int kt = 0; kt < KT2; ++kt) {
    const v16bf a = frag_from(
        __builtin_bit_cast(v8bf, Hlds[(kt * 32 + lane) * 2 + 0]),
        __builtin_bit_cast(v8bf, Hlds[(kt * 32 + lane) * 2 + 1]));
    const uint4* bp = (const uint4*)(W2frag +
        ((long)(kt * NT2 + wv) * 32 + lane) * 16);
    if (kt + 1 < KT2)
      __builtin_prefetch(W2frag + ((long)((kt + 1) * NT2 + wv) * 32 + lane) * 16, 0, 1);
    const v16bf b = frag_from(__builtin_bit_cast(v8bf, bp[0]),
                              __builtin_bit_cast(v8bf, bp[1]));
    acc = __builtin_amdgcn_wmma_f32_16x16x32_bf16(
        false, a, false, b, (short)0, acc, false, false);
  }

  const float bias = b2[ncol];
#pragma unroll
  for (int r = 0; r < 8; ++r) {
    const long row = (long)mtile * 16 + r + 8 * half;
    out[row * OUTD + ncol] = acc[r] + bias;
  }
}

// ----------------------------------------------------------------- launch --
extern "C" void kernel_launch(void* const* d_in, const int* in_sizes, int n_in,
                              void* d_out, int out_size, void* d_ws, size_t ws_size,
                              hipStream_t stream) {
  const float* news = (const float*)d_in[0];
  const float* entf = (const float*)d_in[1];
  const float* topf = (const float*)d_in[2];
  const int*   erow = (const int*)d_in[3];
  const int*   ecol = (const int*)d_in[4];
  const int*   trow = (const int*)d_in[5];
  const int*   tcol = (const int*)d_in[6];
  const float* W1   = (const float*)d_in[7];
  const float* b1   = (const float*)d_in[8];
  const float* W2   = (const float*)d_in[9];
  const float* b2   = (const float*)d_in[10];
  float* out = (float*)d_out;
  const int nE = in_sizes[3];
  const int nT = in_sizes[5];

  // Workspace layout (bytes):
  //   [0,            102,400,000) agg    f32 [N_NEWS][256] (ent|top sums)
  //   [102,400,000,  102,800,000) dege   f32 [N_NEWS]
  //   [102,800,000,  103,200,000) degt   f32 [N_NEWS]
  //   [103,200,000,  205,600,000) H16    bf16 [N_NEWS][512]
  //   [205,600,000,  205,993,216) W1frag bf16 [12][32][32][16]
  //   [205,993,216,  206,124,288) W2frag bf16 [16][8][32][16]
  char* ws = (char*)d_ws;
  float*  agg    = (float*)(ws);
  float*  dege   = (float*)(ws + 102400000L);
  float*  degt   = (float*)(ws + 102800000L);
  __bf16* H16    = (__bf16*)(ws + 103200000L);
  __bf16* W1frag = (__bf16*)(ws + 205600000L);
  __bf16* W2frag = (__bf16*)(ws + 205993216L);

  // Zero agg + both degree arrays (contiguous 103,200,000 B = 6,450,000 f4).
  zero_f4<<<2048, 256, 0, stream>>>((float4*)ws, 103200000L / 16);

  // Pack weights into WMMA B-fragment bf16 layout (one-time, tiny).
  make_wfrag<<<(KT1 * NT1 * 32 + 255) / 256, 256, 0, stream>>>(
      W1, W1frag, HID, NT1, KT1 * NT1 * 32);
  make_wfrag<<<(KT2 * NT2 * 32 + 255) / 256, 256, 0, stream>>>(
      W2, W2frag, OUTD, NT2, KT2 * NT2 * 32);

  edge_agg<<<4096, 256, 0, stream>>>(entf, erow, ecol, agg, dege, nE, 0);
  edge_agg<<<4096, 256, 0, stream>>>(topf, trow, tcol, agg, degt, nT, 128);

  normalize_agg<<<2048, 256, 0, stream>>>(agg, dege, degt);

  gemm1_tanh<<<N_NEWS / 16, 256, 0, stream>>>(news, agg, W1frag, b1, H16);
  gemm2_bias<<<N_NEWS / 16, 256, 0, stream>>>(H16, W2frag, b2, out);
}
